// ClosedFlyLoop_16844861735014
// MI455X (gfx1250) — matrix operand order repaired and checked by reading, not compiled, not run on hardware
//
#include <hip/hip_runtime.h>

typedef __attribute__((ext_vector_type(16))) _Float16 v16h;
typedef __attribute__((ext_vector_type(8)))  _Float16 v8h;
typedef __attribute__((ext_vector_type(8)))  float    v8f;

#define RAD 12          // truncate*sigma+0.5 = 12 for sigma=3
#define KW  40          // 16 + 2*RAD  (Toeplitz K dimension)
#define AP_CUT 15
#define TPW 4           // 16x16 tiles per wave along x (amortize weight build)
#define YSTR 68         // Y-pass LDS row stride in halves (34 dwords: conflict-free)
#define XSTR 88         // X-pass LDS row stride in halves (64 + 24 halo)

// ---- Gaussian kernel weights (computed in-register, no LUT -> no scratch) ----
__device__ __forceinline__ float kern_inv() {
  float s = 0.0f;
#pragma unroll
  for (int n = -RAD; n <= RAD; ++n)
    s += __expf((float)(n * n) * (-1.0f / 18.0f));   // sigma=3 -> -x^2/(2*9)
  return 1.0f / s;
}
__device__ __forceinline__ float gw(int d, float inv) {  // d in [0,24] else 0
  if ((unsigned)d > 24u) return 0.0f;
  float t = (float)(d - RAD);
  return inv * __expf(t * t * (-1.0f / 18.0f));
}

// 16-bit A-matrix 16x32 lane/element -> K map (ISA 7.12.2)
__device__ __forceinline__ int a_k(int lane, int e) {
  int base = (lane & 16) ? 8 : 0;
  return (e < 8) ? (base + e) : (base + 8 + e);
}

// =====================  Y-direction smooth (circular)  =====================
// Per wave: one 16-row x 64-col output strip. 40x64 halo patch staged in LDS
// via coalesced 8B loads; Toeplitz weights in A built once, 4 tiles x 2 WMMA.
__global__ void __launch_bounds__(256)
smooth_y_kernel(const _Float16* __restrict__ in, _Float16* __restrict__ out,
                int C, int Y, int X) {
  __shared__ __align__(16) _Float16 lds[8 * 40 * YSTR];   // 43.5 KB / block
  const int lane  = threadIdx.x & 31;
  const int wslot = threadIdx.x >> 5;
  const int wave  = (int)((blockIdx.x * blockDim.x + threadIdx.x) >> 5);
  const int gx = X >> 6, gy = Y >> 4;
  const int perField = gx * gy;
  const bool valid = wave < C * perField;
  int f = 0, y0 = 0, wx0 = 0;
  if (valid) {
    f = wave / perField;
    int t = wave - f * perField;
    y0  = (t / gx) << 4;
    wx0 = (t % gx) << 6;
  }
  _Float16* patch = lds + wslot * (40 * YSTR);
  if (valid) {
    const _Float16* base = in + (size_t)f * Y * X;
    for (int i = lane; i < 40 * 16; i += 32) {       // 8-byte chunks, coalesced
      int r  = i >> 4;
      int cc = (i & 15) << 2;
      int row = y0 - RAD + r; row %= Y; if (row < 0) row += Y;   // circular
      *(unsigned long long*)(patch + r * YSTR + cc) =
          *(const unsigned long long*)(base + (size_t)row * X + wx0 + cc);
    }
  }
  __syncthreads();
  if (!valid) return;

  const float inv = kern_inv();
  const int m = lane & 15;
  // A = banded Toeplitz weights: A[m][k] = kern[k-m] for 0<=k-m<=24 (built ONCE)
  v16h a0, a1;
#pragma unroll
  for (int e = 0; e < 16; ++e) {
    int k = a_k(lane, e);
    a0[e] = (_Float16)gw(k - m, inv);
    a1[e] = (_Float16)gw(32 + k - m, inv);
  }

  const int n = lane & 15;
  const int koff = (lane & 16) ? 16 : 0;
  const int mh   = (lane & 16) ? 8 : 0;
  _Float16* ob = out + (size_t)f * Y * X;

#pragma unroll
  for (int tx = 0; tx < TPW; ++tx) {
    const int cbase = (tx << 4) + n;
    v16h b0, b1;                                    // B = data column gathers (LDS)
#pragma unroll
    for (int e = 0; e < 16; ++e) {
      int k = a_k(lane, e);
      b0[e] = patch[k * YSTR + cbase];
      // Chunk-1: A1 weights are zero for all k>=8 (32+k-m >= 25), so values at
      // clamped rows are multiplied by 0 -> branchless clamp, no divergent EXEC.
      int k1 = 32 + k; k1 = (k1 > KW - 1) ? (KW - 1) : k1;
      b1[e] = patch[k1 * YSTR + cbase];
    }
    v8f c = {};
    c = __builtin_amdgcn_wmma_f32_16x16x32_f16(false, a0, false, b0, (short)0, c, false, false);
    c = __builtin_amdgcn_wmma_f32_16x16x32_f16(false, a1, false, b1, (short)0, c, false, false);
#pragma unroll
    for (int r = 0; r < 8; ++r)
      ob[(size_t)(y0 + mh + r) * X + wx0 + cbase] = (_Float16)c[r];
  }
}

// =====================  X-direction smooth (edge-clamp)  =====================
// Per wave: 16-row x 64-col strip; 16x88 patch in LDS. A = data (three 16B
// ds_load_b128 per fragment, bank-perfect); B = Toeplitz weights built once.
template <typename OutT>
__global__ void __launch_bounds__(256)
smooth_x_kernel(const _Float16* __restrict__ in, OutT* __restrict__ out,
                int C, int Y, int X) {
  __shared__ __align__(16) _Float16 lds[8 * 16 * XSTR];   // 22 KB / block
  const int lane  = threadIdx.x & 31;
  const int wslot = threadIdx.x >> 5;
  const int wave  = (int)((blockIdx.x * blockDim.x + threadIdx.x) >> 5);
  const int gx = X >> 6, gy = Y >> 4;
  const int perField = gx * gy;
  const bool valid = wave < C * perField;
  int f = 0, y0 = 0, wx0 = 0;
  if (valid) {
    f = wave / perField;
    int t = wave - f * perField;
    y0  = (t / gx) << 4;
    wx0 = (t % gx) << 6;
  }
  _Float16* patch = lds + wslot * (16 * XSTR);
  if (valid) {
    const _Float16* base = in + (size_t)f * Y * X + (size_t)y0 * X;
    if (wx0 != 0 && wx0 != X - 64) {                // interior: vector copy
      const _Float16* src0 = base + (wx0 - RAD);
      for (int i = lane; i < 16 * 22; i += 32) {    // 88 cols = 22 x 8B chunks
        int r = i / 22, cc = (i - r * 22) << 2;
        *(unsigned long long*)(patch + r * XSTR + cc) =
            *(const unsigned long long*)(src0 + (size_t)r * X + cc);
      }
    } else {                                        // edge: scalar with clamp
      for (int i = lane; i < 16 * XSTR; i += 32) {
        int r = i / XSTR, cc = i - r * XSTR;
        int col = wx0 - RAD + cc;
        col = col < 0 ? 0 : (col > X - 1 ? X - 1 : col);
        patch[r * XSTR + cc] = base[(size_t)r * X + col];
      }
    }
  }
  __syncthreads();
  if (!valid) return;

  const float inv = kern_inv();
  // B = banded Toeplitz: B[k][n] = kern[k-n]  (built ONCE per wave)
  const int n = lane & 15;
  const int koff = (lane & 16) ? 16 : 0;
  v16h b0, b1;
#pragma unroll
  for (int e = 0; e < 16; ++e) {
    int k = a_k(lane, e);
    b0[e] = (_Float16)gw(k - n, inv);
    b1[e] = (_Float16)gw(32 + k - n, inv);
  }

  const int m = lane & 15;
  const int base8 = (lane & 16) ? 8 : 0;
  const int mh    = (lane & 16) ? 8 : 0;
  OutT* ob = out + (size_t)f * Y * X;

#pragma unroll
  for (int tx = 0; tx < TPW; ++tx) {
    const _Float16* ap = patch + m * XSTR + (tx << 4);
    v8h r0 = *(const v8h*)(ap + base8);             // k = base8 .. base8+7
    v8h r1 = *(const v8h*)(ap + base8 + 16);        // k = base8+16 .. +23
    v8h r2 = *(const v8h*)(ap + 32);                // k1 = 32..39 (chunk 1)
    v16h a0, a1;
    // Chunk-1 positions with k1 >= 40 carry zero B1 weights (32+k-n >= 25),
    // so no masking is needed: stale data there contributes 0 to the matmul.
#pragma unroll
    for (int e = 0; e < 8; ++e) {
      a0[e]     = r0[e];
      a0[e + 8] = r1[e];
      a1[e]     = r2[e];
      a1[e + 8] = r2[e];
    }
    v8f c = {};
    c = __builtin_amdgcn_wmma_f32_16x16x32_f16(false, a0, false, b0, (short)0, c, false, false);
    c = __builtin_amdgcn_wmma_f32_16x16x32_f16(false, a1, false, b1, (short)0, c, false, false);
#pragma unroll
    for (int r = 0; r < 8; ++r)
      ob[(size_t)(y0 + mh + r) * X + wx0 + (tx << 4) + n] = (OutT)c[r];
  }
}

// =====================  prep: f32 -> f16, LR-symmetrize v  =====================
__global__ void __launch_bounds__(256)
prep_kernel(const float* __restrict__ yraw, const float* __restrict__ vraw,
            _Float16* __restrict__ F, int Y, int X) {
  size_t plane = (size_t)Y * X;
  size_t idx = (size_t)blockIdx.x * blockDim.x + threadIdx.x;
  if (idx >= plane) return;
  int yy = (int)(idx / X), xx = (int)(idx - (size_t)yy * X);
#pragma unroll
  for (int f = 0; f < 5; ++f)
    F[f * plane + idx] = (_Float16)yraw[f * plane + idx];
  size_t fidx = (size_t)(Y - 1 - yy) * X + xx;            // flip along DV (Y)
  F[5 * plane + idx] = (_Float16)(0.5f * (vraw[idx]         - vraw[fidx]));
  F[6 * plane + idx] = (_Float16)(0.5f * (vraw[plane + idx] + vraw[plane + fidx]));
}

// ===========  gradients (central diff) + tensor algebra + AP cut  ===========
__global__ void __launch_bounds__(256)
grad_algebra_kernel(const _Float16* __restrict__ G,  // 7 smoothed f16 fields
                    const float* __restrict__ yraw,  // raw state (m, s)
                    const float* __restrict__ vraw,  // raw velocity
                    _Float16* __restrict__ H,        // out: 5 f16 fields
                    int Y, int X) {
  size_t plane = (size_t)Y * X;
  size_t idx = (size_t)blockIdx.x * blockDim.x + threadIdx.x;
  if (idx >= plane) return;
  int yy = (int)(idx / X), xx = (int)(idx - (size_t)yy * X);
  int ym = (yy == 0)     ? Y - 1 : yy - 1;   // circular in Y
  int yp = (yy == Y - 1) ? 0     : yy + 1;
  int xm = (xx == 0)     ? 0     : xx - 1;   // replicate in X
  int xp = (xx == X - 1) ? X - 1 : xx + 1;

  auto dYf = [&](int f) -> float {
    return 0.5f * ((float)G[(size_t)f * plane + (size_t)yp * X + xx] -
                   (float)G[(size_t)f * plane + (size_t)ym * X + xx]);
  };
  auto dXf = [&](int f) -> float {
    return 0.5f * ((float)G[(size_t)f * plane + (size_t)yy * X + xp] -
                   (float)G[(size_t)f * plane + (size_t)yy * X + xm]);
  };

  float mm[2][2] = {{yraw[0 * plane + idx], yraw[1 * plane + idx]},
                    {yraw[2 * plane + idx], yraw[3 * plane + idx]}};
  float s = yraw[4 * plane + idx];
  size_t fidx = (size_t)(Y - 1 - yy) * X + xx;
  float v0 = 0.5f * (vraw[idx]         - vraw[fidx]);
  float v1 = 0.5f * (vraw[plane + idx] + vraw[plane + fidx]);

  float sdot = -(v0 * dYf(4) + v1 * dXf(4));

  // T[i][j] = d_j (smoothed v_i);  fields 5,6
  float T01 = dXf(5), T10 = dYf(6);
  float E00 = dYf(5), E11 = dXf(6);          // diag(E) == diag(T)
  float O01 = -0.5f * (T01 - T10);           // vorticity
  float O[2][2] = {{0.0f, O01}, {-O01, 0.0f}};
  float trE = E00 + E11;
  float trm = mm[0][0] + mm[1][1];
  float cm = -(0.11f - 0.099f * s) + (0.767f + 0.055f * s) * trE
           + (0.732f - 0.59f * s) * trm;
  float cg = (0.069f - 0.048f * s) * trm;    // gamma_dv = diag(1,0)

  bool cut = (xx < AP_CUT) || (xx >= X - AP_CUT);
  float outm[2][2];
#pragma unroll
  for (int i = 0; i < 2; ++i)
#pragma unroll
    for (int j = 0; j < 2; ++j) {
      int fij = i * 2 + j;
      float lhs = v0 * dYf(fij) + v1 * dXf(fij);
      lhs += O[i][0] * mm[0][j] + O[i][1] * mm[1][j];
      lhs -= mm[i][0] * O[0][j] + mm[i][1] * O[1][j];
      float rhs = cm * mm[i][j] + ((i == 0 && j == 0) ? cg : 0.0f);
      outm[i][j] = cut ? 0.0f : (-lhs + rhs);
    }
  H[0 * plane + idx] = (_Float16)outm[0][0];
  H[1 * plane + idx] = (_Float16)outm[0][1];
  H[2 * plane + idx] = (_Float16)outm[1][0];
  H[3 * plane + idx] = (_Float16)outm[1][1];
  H[4 * plane + idx] = (_Float16)(cut ? 0.0f : sdot);
}

// ===============================  launcher  ===============================
extern "C" void kernel_launch(void* const* d_in, const int* in_sizes, int n_in,
                              void* d_out, int out_size, void* d_ws, size_t ws_size,
                              hipStream_t stream) {
  (void)in_sizes; (void)n_in; (void)out_size; (void)ws_size;
  const int Y = 1024, X = 2048;
  const float* yraw = (const float*)d_in[0];   // [5,Y,X] f32
  const float* vraw = (const float*)d_in[1];   // [2,Y,X] f32
  float* outp = (float*)d_out;                 // [5,Y,X] f32

  size_t plane = (size_t)Y * X;
  _Float16* buf0 = (_Float16*)d_ws;                                  // 7 fields
  _Float16* buf1 = (_Float16*)((char*)d_ws + 7 * plane * sizeof(_Float16));

  const int threads = 256;
  const int pxBlocks = (int)((plane + threads - 1) / threads);
  const int g7 = 7 * (Y / 16) * (X / 64);      // wave-strips (16 rows x 64 cols)
  const int g5 = 5 * (Y / 16) * (X / 64);
  const int b7 = (g7 * 32 + threads - 1) / threads;
  const int b5 = (g5 * 32 + threads - 1) / threads;

  // 1. prep: raw -> f16 fields {m00,m01,m10,m11,s,v0sym,v1sym}
  prep_kernel<<<pxBlocks, threads, 0, stream>>>(yraw, vraw, buf0, Y, X);
  // 2-3. separable Gaussian smooth of the 7 fields (WMMA Toeplitz matmuls)
  smooth_y_kernel<<<b7, threads, 0, stream>>>(buf0, buf1, 7, Y, X);
  smooth_x_kernel<_Float16><<<b7, threads, 0, stream>>>(buf1, buf0, 7, Y, X);
  // 4. gradients + einsum algebra + AP cut -> 5 f16 fields
  grad_algebra_kernel<<<pxBlocks, threads, 0, stream>>>(buf0, yraw, vraw, buf1, Y, X);
  // 5-6. postprocess smooth, final X pass writes f32 directly to d_out
  smooth_y_kernel<<<b5, threads, 0, stream>>>(buf1, buf0, 5, Y, X);
  smooth_x_kernel<float><<<b5, threads, 0, stream>>>(buf0, outp, 5, Y, X);
}
